// DeformableConv2DLayer_90615220011294
// MI455X (gfx1250) — matrix operand order repaired and checked by reading, not compiled, not run on hardware
//
#include <hip/hip_runtime.h>
#include <hip/hip_bf16.h>

typedef __attribute__((ext_vector_type(16))) _Float16 v16h;
typedef __attribute__((ext_vector_type(8)))  _Float16 v8h;
typedef __attribute__((ext_vector_type(8)))  float    v8f;

#define BN    4
#define CIN   64
#define HN    128
#define WN    128
#define COUT  64
#define HW    (HN * WN)        // 16384
#define KTAPS 9
#define KDIM  (CIN * KTAPS)    // 576 contraction length, ordered k = tap*64 + c

struct Param {                 // 32B, 32B-aligned record per (b, h, w, tap)
    int4   idx;                // 4 clamped corner pixel indices (y*W + x)
    float4 w;                  // 4 bilinear weights, validity & sigmoid-mask folded in
};

// ---------------------------------------------------------------------------
// Kernel A: NCHW f32 -> NHWC f16 (channel-contiguous for wide corner gathers)
// ---------------------------------------------------------------------------
__global__ __launch_bounds__(256) void k_transpose(const float* __restrict__ x,
                                                   _Float16* __restrict__ xt) {
    int idx = blockIdx.x * 256 + threadIdx.x;       // (b, h, w) flat
    int w = idx & (WN - 1);
    int h = (idx >> 7) & (HN - 1);
    int b = idx >> 14;
    v16h v0, v1, v2, v3;
#pragma unroll
    for (int c = 0; c < CIN; ++c) {
        // coalesced across lanes (consecutive w) for each channel plane
        float f = x[(((size_t)b * CIN + c) * HN + h) * WN + w];
        _Float16 hf = (_Float16)f;
        if (c < 16)      v0[c]      = hf;
        else if (c < 32) v1[c - 16] = hf;
        else if (c < 48) v2[c - 32] = hf;
        else             v3[c - 48] = hf;
    }
    v16h* dst = (v16h*)(xt + (size_t)idx * CIN);    // 128B per thread, contiguous
    dst[0] = v0; dst[1] = v1; dst[2] = v2; dst[3] = v3;
}

// ---------------------------------------------------------------------------
// Kernel B: offset/mask conv (27 ch) + precompute bilinear sampling params
// ---------------------------------------------------------------------------
__global__ __launch_bounds__(256) void k_offset(const float* __restrict__ x,
                                                const float* __restrict__ ow,
                                                const float* __restrict__ ob,
                                                Param* __restrict__ params) {
    __shared__ float sw[27 * CIN * KTAPS];          // 62208 B
    for (int i = threadIdx.x; i < 27 * CIN * KTAPS; i += 256) sw[i] = ow[i];
    __syncthreads();

    int idx = blockIdx.x * 256 + threadIdx.x;       // (b, h, w) flat
    int w = idx & (WN - 1);
    int h = (idx >> 7) & (HN - 1);
    int b = idx >> 14;

    float acc[27];
#pragma unroll
    for (int o = 0; o < 27; ++o) acc[o] = ob[o];

    for (int c = 0; c < CIN; ++c) {
        float xv[9];
#pragma unroll
        for (int kh = 0; kh < 3; ++kh) {
            int yy = h - 1 + kh;
#pragma unroll
            for (int kw = 0; kw < 3; ++kw) {
                int xx = w - 1 + kw;
                bool vok = (yy >= 0) && (yy < HN) && (xx >= 0) && (xx < WN);
                xv[kh * 3 + kw] = vok ? x[(((size_t)b * CIN + c) * HN + yy) * WN + xx] : 0.0f;
            }
        }
#pragma unroll
        for (int kk = 0; kk < 9; ++kk) {
            float v = xv[kk];
#pragma unroll
            for (int o = 0; o < 27; ++o)            // LDS reads are lane-uniform (broadcast)
                acc[o] = fmaf(v, sw[(o * CIN + c) * 9 + kk], acc[o]);
        }
    }

#pragma unroll
    for (int t = 0; t < KTAPS; ++t) {
        float dy = acc[t];
        float dx = acc[9 + t];
        float ms = 1.0f / (1.0f + expf(-acc[18 + t]));
        float py = (float)(h - 1 + t / 3) + dy;
        float px = (float)(w - 1 + t % 3) + dx;
        float fy = floorf(py), fx = floorf(px);
        float ly = py - fy,    lx = px - fx;
        int y0 = (int)fy, x0 = (int)fx;
        int y1 = y0 + 1,  x1 = x0 + 1;

        bool vy0 = (y0 >= 0) && (y0 < HN), vy1 = (y1 >= 0) && (y1 < HN);
        bool vx0 = (x0 >= 0) && (x0 < WN), vx1 = (x1 >= 0) && (x1 < WN);
        int cy0 = min(max(y0, 0), HN - 1), cy1 = min(max(y1, 0), HN - 1);
        int cx0 = min(max(x0, 0), WN - 1), cx1 = min(max(x1, 0), WN - 1);

        Param p;
        p.idx = make_int4(cy0 * WN + cx0, cy0 * WN + cx1,
                          cy1 * WN + cx0, cy1 * WN + cx1);
        p.w   = make_float4((vy0 && vx0) ? (1.0f - ly) * (1.0f - lx) * ms : 0.0f,
                            (vy0 && vx1) ? (1.0f - ly) * lx          * ms : 0.0f,
                            (vy1 && vx0) ? ly          * (1.0f - lx) * ms : 0.0f,
                            (vy1 && vx1) ? ly          * lx          * ms : 0.0f);
        params[(size_t)idx * KTAPS + t] = p;
    }
}

// ---------------------------------------------------------------------------
// Kernel C: fused deformable-im2col + GEMM via v_wmma_f32_16x16x32_f16
//   block = (b, output row h): 8 waves x 16 positions = 128 positions
//   wave  = 16 positions x full Cout=64 (4 M-tiles), K loop = 9 taps x 2 chunks
// ---------------------------------------------------------------------------
static __device__ inline v16h splat16(_Float16 s) {
    v16h v;
#pragma unroll
    for (int i = 0; i < 16; ++i) v[i] = s;
    return v;
}

__global__ __launch_bounds__(256) void k_dcn_gemm(const float* __restrict__ weight,
                                                  const float* __restrict__ bias,
                                                  const _Float16* __restrict__ xt,
                                                  const Param* __restrict__ params,
                                                  float* __restrict__ out) {
    extern __shared__ _Float16 wlds[];              // [COUT][KDIM], K = tap*64 + c : 72KB

    for (int i = threadIdx.x; i < COUT * KDIM; i += 256) {
        int o = i / KDIM, k = i % KDIM;
        int t = k >> 6, c = k & 63;
        wlds[o * KDIM + k] = (_Float16)weight[((size_t)o * CIN + c) * KTAPS + t];
    }
    __syncthreads();

    int h = blockIdx.x & (HN - 1);
    int b = blockIdx.x >> 7;
    int lane = threadIdx.x & 31;
    int wv   = threadIdx.x >> 5;
    int n    = lane & 15;                           // matrix column (position) / A row
    int hi   = lane >> 4;                           // half-wave select
    int wcol = wv * 16 + n;                         // output x coordinate
    int pos  = h * WN + wcol;

    v8f acc[4] = {v8f{}, v8f{}, v8f{}, v8f{}};

    const Param* pp = params + ((size_t)b * HW + pos) * KTAPS;
    const _Float16* xb = xt + (size_t)b * HW * CIN;
    int kb_a = hi * 8;                              // unused; the A fragment also uses kb_b
    int kb_b = hi * 16;                             // B-fragment channel base within chunk

#pragma unroll 1
    for (int t = 0; t < KTAPS; ++t) {
        Param p = pp[t];
        _Float16 w00 = (_Float16)p.w.x, w01 = (_Float16)p.w.y;
        _Float16 w10 = (_Float16)p.w.z, w11 = (_Float16)p.w.w;
        v16h s00 = splat16(w00), s01 = splat16(w01);
        v16h s10 = splat16(w10), s11 = splat16(w11);

#pragma unroll
        for (int hc = 0; hc < 2; ++hc) {
            int coff = hc * 32 + kb_b;              // 16 contiguous channels per lane
            v16h c00 = *(const v16h*)(xb + (size_t)p.idx.x * CIN + coff);
            v16h c01 = *(const v16h*)(xb + (size_t)p.idx.y * CIN + coff);
            v16h c10 = *(const v16h*)(xb + (size_t)p.idx.z * CIN + coff);
            v16h c11 = *(const v16h*)(xb + (size_t)p.idx.w * CIN + coff);

            // modulated bilinear blend -> B fragment (v_pk_fma_f16)
            v16h bfrag = s00 * c00;
            bfrag += s01 * c01;
            bfrag += s10 * c10;
            bfrag += s11 * c11;

            int kg = t * 64 + hc * 32;
#pragma unroll
            for (int m = 0; m < 4; ++m) {
                const _Float16* wrow = wlds + (size_t)(m * 16 + n) * KDIM + kg;
                v8h lo  = *(const v8h*)(wrow + kb_b);        // K = kb_b .. kb_b+7
                v8h hi8 = *(const v8h*)(wrow + kb_b + 8);    // K = kb_b+8 .. kb_b+15
                v16h afrag = __builtin_shufflevector(lo, hi8,
                    0, 1, 2, 3, 4, 5, 6, 7, 8, 9, 10, 11, 12, 13, 14, 15);
                acc[m] = __builtin_amdgcn_wmma_f32_16x16x32_f16(
                    false, afrag, false, bfrag, (short)0, acc[m], false, false);
            }
        }
    }

    // Epilogue: C/D layout -> VGPR r holds M = r + 8*hi, column N = n
#pragma unroll
    for (int m = 0; m < 4; ++m) {
#pragma unroll
        for (int r = 0; r < 8; ++r) {
            int co = m * 16 + r + hi * 8;
            out[((size_t)b * COUT + co) * HW + pos] = acc[m][r] + bias[co];
        }
    }
}

// ---------------------------------------------------------------------------
extern "C" void kernel_launch(void* const* d_in, const int* in_sizes, int n_in,
                              void* d_out, int out_size, void* d_ws, size_t ws_size,
                              hipStream_t stream) {
    (void)in_sizes; (void)n_in; (void)out_size; (void)ws_size;
    const float* x      = (const float*)d_in[0];
    const float* weight = (const float*)d_in[1];
    const float* bias   = (const float*)d_in[2];
    const float* ow     = (const float*)d_in[3];
    const float* ob     = (const float*)d_in[4];
    float* out = (float*)d_out;

    _Float16* xt   = (_Float16*)d_ws;                                   // 8 MB
    Param*    prm  = (Param*)((char*)d_ws + (size_t)BN * HW * CIN * 2); // ~19 MB

    k_transpose<<<BN * HW / 256, 256, 0, stream>>>(x, xt);
    k_offset   <<<BN * HW / 256, 256, 0, stream>>>(x, ow, ob, prm);
    k_dcn_gemm <<<BN * HN, 256, COUT * KDIM * sizeof(_Float16), stream>>>(
        weight, bias, xt, prm, out);
}
